// B_MLUTNet_45140106281099
// MI455X (gfx1250) — 8-bit matrix operand order repaired and checked by reading, compiled, not run on hardware
//
#include <hip/hip_runtime.h>
#include <cstdint>

#define D_IN   1024
#define HDIM   4096
#define D_OUTV 1000
#define NPAD56 1024
#define BATCH  8192
#define EPSV   1e-5f

typedef __attribute__((ext_vector_type(8))) int v8i;

__device__ __forceinline__ int8_t fsign(float v) {
  return v > 0.f ? (int8_t)1 : (v < 0.f ? (int8_t)-1 : (int8_t)0);
}


__global__ void k_pack_pad(const float* __restrict__ src, int8_t* __restrict__ dst,
                           int rows_valid, int rows_pad, int cols) {
  long i = (long)blockIdx.x * blockDim.x + threadIdx.x;
  long total = (long)rows_pad * cols;
  if (i >= total) return;
  int r = (int)(i / cols);
  dst[i] = (r < rows_valid) ? fsign(src[i]) : (int8_t)0;
}

__global__ void k_pack_x(const float* __restrict__ x,
                         int8_t* __restrict__ p1, int8_t* __restrict__ p2) {
  long i = (long)blockIdx.x * blockDim.x + threadIdx.x;
  if (i >= (long)BATCH * D_IN) return;
  int r = (int)(i / D_IN), c = (int)(i % D_IN);
  const float* row = x + (long)r * (2 * D_IN);
  p1[i] = fsign(row[c]);
  p2[i] = fsign(row[c + D_IN]);
}

__global__ void k_pack_sign(const float* __restrict__ a, int8_t* __restrict__ dst, long n) {
  long i = (long)blockIdx.x * blockDim.x + threadIdx.x;
  if (i < n) dst[i] = fsign(a[i]);
}

__global__ void k_pack_add(const float* __restrict__ a, const float* __restrict__ b,
                           int8_t* __restrict__ dst, long n) {
  long i = (long)blockIdx.x * blockDim.x + threadIdx.x;
  if (i < n) dst[i] = fsign(a[i] + b[i]);
}

__device__ __forceinline__ v8i load_fragA(const int8_t* __restrict__ base, int hi) {
  const int4 lo = *(const int4*)(base + hi * 16);
  const int4 up = *(const int4*)(base + 32 + hi * 16);
  v8i f;
  f[0] = lo.x; f[1] = lo.y; f[2] = lo.z; f[3] = lo.w;
  f[4] = up.x; f[5] = up.y; f[6] = up.z; f[7] = up.w;
  return f;
}

__device__ __forceinline__ v8i load_fragB(const int8_t* __restrict__ base, int hi) {
  const int4 lo = *(const int4*)(base + hi * 16);
  const int4 up = *(const int4*)(base + 32 + hi * 16);
  v8i f;
  f[0] = lo.x; f[1] = lo.y; f[2] = lo.z; f[3] = lo.w;
  f[4] = up.x; f[5] = up.y; f[6] = up.z; f[7] = up.w;
  return f;
}

__global__ __launch_bounds__(256) void k_gemm_bin(
    const int8_t* __restrict__ A, const int8_t* __restrict__ W,
    const float* __restrict__ bias, float* __restrict__ C,
    int K, int Nvalid, int ldc, int coff) {
  const int lane = threadIdx.x & 31;
  const int wave = threadIdx.x >> 5;
  const int wm = wave >> 2;
  const int wn = wave & 3;
  const int hi = lane >> 4;
  const int lr = lane & 15;

  const int m0 = blockIdx.y * 64 + wm * 32;
  const int n0 = blockIdx.x * 128 + wn * 32;

  v8i acc[2][2] = {};

  const int8_t* Arow0 = A + (long)(m0 + lr) * K;
  const int8_t* Arow1 = A + (long)(m0 + 16 + lr) * K;
  const int8_t* Wrow0 = W + (long)(n0 + lr) * K;
  const int8_t* Wrow1 = W + (long)(n0 + 16 + lr) * K;

  for (int k = 0; k < K; k += 64) {
    v8i a0 = load_fragA(Arow0 + k, hi);
    v8i a1 = load_fragA(Arow1 + k, hi);
    v8i b0 = load_fragB(Wrow0 + k, hi);
    v8i b1 = load_fragB(Wrow1 + k, hi);
    acc[0][0] = __builtin_amdgcn_wmma_i32_16x16x64_iu8(true, a0, true, b0, acc[0][0], false, false);
    acc[0][1] = __builtin_amdgcn_wmma_i32_16x16x64_iu8(true, a0, true, b1, acc[0][1], false, false);
    acc[1][0] = __builtin_amdgcn_wmma_i32_16x16x64_iu8(true, a1, true, b0, acc[1][0], false, false);
    acc[1][1] = __builtin_amdgcn_wmma_i32_16x16x64_iu8(true, a1, true, b1, acc[1][1], false, false);
  }

#pragma unroll
  for (int mt = 0; mt < 2; ++mt) {
#pragma unroll
    for (int nt = 0; nt < 2; ++nt) {
      int col = n0 + nt * 16 + lr;
      if (col >= Nvalid) continue;
      float bv = bias[col];
      int rowb = m0 + mt * 16 + hi * 8;
#pragma unroll
      for (int j = 0; j < 8; ++j) {
        C[(long)(rowb + j) * ldc + coff + col] = (float)acc[mt][nt][j] + bv;
      }
    }
  }
}


__global__ void k_zero2(float* s, float* q, int n) {
  int i = blockIdx.x * blockDim.x + threadIdx.x;
  if (i < n) { s[i] = 0.f; q[i] = 0.f; }
}

__global__ void k_colstat_partial(const float* __restrict__ h, float* __restrict__ s,
                                  float* __restrict__ q, int N, int rows_per) {
  int col = blockIdx.x * blockDim.x + threadIdx.x;
  if (col >= N) return;
  long base = (long)blockIdx.y * rows_per * N + col;
  float ss = 0.f, qq = 0.f;
  for (int r = 0; r < rows_per; ++r) {
    float v = h[base + (long)r * N];
    ss += v; qq += v * v;
  }
  atomicAdd(&s[col], ss);
  atomicAdd(&q[col], qq);
}

__global__ void k_colstat_final(const float* __restrict__ s, const float* __restrict__ q,
                                float* __restrict__ mu, float* __restrict__ rstd, int N) {
  int c = blockIdx.x * blockDim.x + threadIdx.x;
  if (c >= N) return;
  float m = s[c] / (float)BATCH;
  float var = q[c] / (float)BATCH - m * m;
  mu[c] = m;
  rstd[c] = rsqrtf(var + EPSV);
}

__global__ void k_bnclip(float* __restrict__ h, const float* __restrict__ mu,
                         const float* __restrict__ rstd, const float* __restrict__ g,
                         const float* __restrict__ bt, int N) {
  long i = (long)blockIdx.x * blockDim.x + threadIdx.x;
  if (i >= (long)BATCH * N) return;
  int c = (int)(i % N);
  float v = g[c] * (h[i] - mu[c]) * rstd[c] + bt[c];
  h[i] = fminf(1.f, fmaxf(-1.f, v));
}


__global__ __launch_bounds__(256) void k_logsoftmax(float* __restrict__ y, int N) {
  __shared__ float red[256];
  float* pr = y + (long)blockIdx.x * N;
  int t = threadIdx.x;

  float mx = -1e30f;
  for (int c = t; c < N; c += 256) mx = fmaxf(mx, pr[c]);
  red[t] = mx; __syncthreads();
  for (int s = 128; s > 0; s >>= 1) { if (t < s) red[t] = fmaxf(red[t], red[t + s]); __syncthreads(); }
  mx = red[0]; __syncthreads();

  float sum = 0.f;
  for (int c = t; c < N; c += 256) sum += expf(pr[c] - mx);
  red[t] = sum; __syncthreads();
  for (int s = 128; s > 0; s >>= 1) { if (t < s) red[t] += red[t + s]; __syncthreads(); }
  float lse = mx + logf(red[0]);

  for (int c = t; c < N; c += 256) pr[c] = pr[c] - lse;
}


extern "C" void kernel_launch(void* const* d_in, const int* in_sizes, int n_in,
                              void* d_out, int out_size, void* d_ws, size_t ws_size,
                              hipStream_t stream) {
  const float* x  = (const float*)d_in[0];
  const float* w1 = (const float*)d_in[1];  const float* b1 = (const float*)d_in[2];
  const float* w2 = (const float*)d_in[3];  const float* b2 = (const float*)d_in[4];
  const float* w3 = (const float*)d_in[5];  const float* b3 = (const float*)d_in[6];
  const float* w4 = (const float*)d_in[7];  const float* b4 = (const float*)d_in[8];
  const float* w5 = (const float*)d_in[9];  const float* b5 = (const float*)d_in[10];
  const float* w6 = (const float*)d_in[11]; const float* b6 = (const float*)d_in[12];
  const float* g1 = (const float*)d_in[13]; const float* bt1 = (const float*)d_in[14];
  const float* g2 = (const float*)d_in[15]; const float* bt2 = (const float*)d_in[16];
  const float* g3 = (const float*)d_in[17]; const float* bt3 = (const float*)d_in[18];
  const float* g4 = (const float*)d_in[19]; const float* bt4 = (const float*)d_in[20];

  uint8_t* ws = (uint8_t*)d_ws;
  size_t off = 0;
  auto alloc = [&](size_t bytes) -> void* {
    void* p = ws + off;
    off += (bytes + 255) & ~(size_t)255;
    return p;
  };

  int8_t* w1p = (int8_t*)alloc((size_t)HDIM * D_IN);
  int8_t* w2p = (int8_t*)alloc((size_t)HDIM * D_IN);
  int8_t* w3p = (int8_t*)alloc((size_t)HDIM * HDIM);
  int8_t* w4p = (int8_t*)alloc((size_t)HDIM * HDIM);
  int8_t* w5p = (int8_t*)alloc((size_t)NPAD56 * HDIM);
  int8_t* w6p = (int8_t*)alloc((size_t)NPAD56 * HDIM);
  int8_t* i1p = (int8_t*)alloc((size_t)BATCH * D_IN);
  int8_t* i2p = (int8_t*)alloc((size_t)BATCH * D_IN);
  int8_t* P   = (int8_t*)alloc((size_t)BATCH * HDIM);
  float*  H1  = (float*)alloc((size_t)BATCH * HDIM * sizeof(float));
  float*  H2  = (float*)alloc((size_t)BATCH * HDIM * sizeof(float));
  float*  H3  = (float*)alloc((size_t)BATCH * HDIM * sizeof(float));
  float*  mu   = (float*)alloc(HDIM * sizeof(float));
  float*  rstd = (float*)alloc(HDIM * sizeof(float));
  float*  sumb = (float*)alloc(HDIM * sizeof(float));
  float*  sqb  = (float*)alloc(HDIM * sizeof(float));

  float* out = (float*)d_out;

  const long nBH = (long)BATCH * HDIM;
  auto blks = [](long n) { return (unsigned)((n + 255) / 256); };

  k_pack_pad<<<blks((long)HDIM * D_IN), 256, 0, stream>>>(w1, w1p, HDIM, HDIM, D_IN);
  k_pack_pad<<<blks((long)HDIM * D_IN), 256, 0, stream>>>(w2, w2p, HDIM, HDIM, D_IN);
  k_pack_pad<<<blks((long)HDIM * HDIM), 256, 0, stream>>>(w3, w3p, HDIM, HDIM, HDIM);
  k_pack_pad<<<blks((long)HDIM * HDIM), 256, 0, stream>>>(w4, w4p, HDIM, HDIM, HDIM);
  k_pack_pad<<<blks((long)NPAD56 * HDIM), 256, 0, stream>>>(w5, w5p, D_OUTV, NPAD56, HDIM);
  k_pack_pad<<<blks((long)NPAD56 * HDIM), 256, 0, stream>>>(w6, w6p, D_OUTV, NPAD56, HDIM);

  k_pack_x<<<blks((long)BATCH * D_IN), 256, 0, stream>>>(x, i1p, i2p);

  auto gemm = [&](const int8_t* A, const int8_t* W, const float* bias, float* C,
                  int K, int Npad, int Nvalid, int ldc, int coff) {
    dim3 grid(Npad / 128, BATCH / 64);
    k_gemm_bin<<<grid, 256, 0, stream>>>(A, W, bias, C, K, Nvalid, ldc, coff);
  };

  auto bn = [&](float* h, const float* g, const float* bt) {
    k_zero2<<<(HDIM + 255) / 256, 256, 0, stream>>>(sumb, sqb, HDIM);
    dim3 gs(HDIM / 256, 16);
    k_colstat_partial<<<gs, 256, 0, stream>>>(h, sumb, sqb, HDIM, BATCH / 16);
    k_colstat_final<<<(HDIM + 255) / 256, 256, 0, stream>>>(sumb, sqb, mu, rstd, HDIM);
    k_bnclip<<<blks(nBH), 256, 0, stream>>>(h, mu, rstd, g, bt, HDIM);
  };

  gemm(i1p, w1p, b1, H1, D_IN, HDIM, HDIM, HDIM, 0);
  bn(H1, g1, bt1);

  gemm(i2p, w2p, b2, H2, D_IN, HDIM, HDIM, HDIM, 0);
  bn(H2, g2, bt2);

  k_pack_sign<<<blks(nBH), 256, 0, stream>>>(H1, P, nBH);
  gemm(P, w3p, b3, H3, HDIM, HDIM, HDIM, HDIM, 0);
  bn(H3, g3, bt3);

  k_pack_add<<<blks(nBH), 256, 0, stream>>>(H1, H2, P, nBH);
  gemm(P, w4p, b4, H2, HDIM, HDIM, HDIM, HDIM, 0);
  bn(H2, g4, bt4);

  k_pack_add<<<blks(nBH), 256, 0, stream>>>(H3, H2, P, nBH);
  gemm(P, w5p, b5, out, HDIM, NPAD56, D_OUTV, 2 * D_OUTV, 0);

  k_pack_sign<<<blks(nBH), 256, 0, stream>>>(H2, P, nBH);
  gemm(P, w6p, b6, out, HDIM, NPAD56, D_OUTV, 2 * D_OUTV, D_OUTV);

  k_logsoftmax<<<BATCH, 256, 0, stream>>>(out, 2 * D_OUTV);
}
